// SOMVAE_14748917694772
// MI455X (gfx1250) — matrix operand order repaired and checked by reading, not compiled, not run on hardware
//
#include <hip/hip_runtime.h>
#include <hip/hip_bf16.h>

typedef __attribute__((ext_vector_type(16))) _Float16 v16h;
typedef __attribute__((ext_vector_type(8)))  float    v8f;
typedef unsigned int u32;

union FragAB { v16h h; u32 u[8]; };
union FragC  { v8f  f; float x[8]; };

// Raw LDS offset of a __shared__ object: low 32 bits of its flat address
// (LDS aperture maps addr[31:0] -> LDS offset within the group allocation).
#define LDS_OFF(p) ((unsigned)(size_t)(const void*)(p))

__device__ __forceinline__ void async_ld_b128(unsigned lds, const void* gaddr) {
    asm volatile("global_load_async_to_lds_b128 %0, %1, off"
                 :: "v"(lds), "v"((unsigned long long)(size_t)gaddr)
                 : "memory");
}
__device__ __forceinline__ void wait_async0() {
    asm volatile("s_wait_asynccnt 0x0" ::: "memory");
}

// ---------------------------------------------------------------------------
// Weight prep kernels (f32 -> f16 with layout changes for WMMA B fragments)
// ---------------------------------------------------------------------------

// conv weights [4,4,ci=256,co=256] f32 -> [tap][co][ci] f16
__global__ __launch_bounds__(256) void k_w_tap_t(const float* __restrict__ w,
                                                 _Float16* __restrict__ wt) {
    int i = blockIdx.x * 256 + threadIdx.x;      // 16*256*256
    int ci = i & 255;
    int co = (i >> 8) & 255;
    int tap = i >> 16;
    wt[i] = (_Float16)w[(tap * 256 + ci) * 256 + co];
}

// enc_w [12544,64] f32 -> [64][12544] f16
__global__ __launch_bounds__(256) void k_enc_wt(const float* __restrict__ w,
                                                _Float16* __restrict__ wt) {
    int i = blockIdx.x * 256 + threadIdx.x;      // 64*12544
    int k = i % 12544;
    int n = i / 12544;
    wt[i] = (_Float16)w[(long)k * 64 + n];
}

// dec_w [64,12544] f32 -> [12544][64] f16
__global__ __launch_bounds__(256) void k_dec_wt(const float* __restrict__ w,
                                                _Float16* __restrict__ wt) {
    int i = blockIdx.x * 256 + threadIdx.x;      // 12544*64
    int k = i & 63;
    int n = i >> 6;
    wt[i] = (_Float16)w[(long)k * 12544 + n];
}

// 256-byte zero pad region used as the async-load source for OOB rows
__global__ void k_zero_pad(_Float16* z) { z[threadIdx.x] = (_Float16)0.f; }

// ---------------------------------------------------------------------------
// conv1 (1 input channel) + relu + 2x2 maxpool, fused. Block = one pooled
// pixel, thread = output channel. Output f16 [B,14,14,256].
// ---------------------------------------------------------------------------
__global__ __launch_bounds__(256) void k_conv1_relu_pool(
    const float* __restrict__ x, const float* __restrict__ w,
    const float* __restrict__ bias, _Float16* __restrict__ out) {
    const int pix = blockIdx.x;                  // b*196 + py*14 + px
    const int b = pix / 196;
    const int rem = pix % 196;
    const int py = rem / 14, px = rem % 14;
    __shared__ float patch[5][5];
    const int t = threadIdx.x;
    if (t < 25) {
        int dy = t / 5, dx = t % 5;
        int iy = 2 * py - 1 + dy, ix = 2 * px - 1 + dx;   // SAME pad_lo = 1
        float v = 0.f;
        if (iy >= 0 && iy < 28 && ix >= 0 && ix < 28)
            v = x[((long)b * 28 + iy) * 28 + ix];
        patch[dy][dx] = v;
    }
    __syncthreads();
    float wr[16];
#pragma unroll
    for (int i = 0; i < 16; ++i) wr[i] = w[i * 256 + t];
    const float bv = bias[t];
    float m = 0.f;                               // relu output >= 0
#pragma unroll
    for (int dy = 0; dy < 2; ++dy)
#pragma unroll
        for (int dx = 0; dx < 2; ++dx) {
            float acc = bv;
#pragma unroll
            for (int ky = 0; ky < 4; ++ky)
#pragma unroll
                for (int kx = 0; kx < 4; ++kx)
                    acc += patch[dy + ky][dx + kx] * wr[ky * 4 + kx];
            m = fmaxf(m, fmaxf(acc, 0.f));
        }
    out[(long)pix * 256 + t] = (_Float16)m;
}

// ---------------------------------------------------------------------------
// Implicit-GEMM 4x4 SAME conv, 256->256 channels, WMMA f16 -> f32.
// `shift`=1: input is 7x7 and virtually 2x-upsampled (deconv1 path).
// Block tile M=64 x N=128; 8 waves = 4(M) x 2(N), each wave 16x64.
// Double-buffered LDS filled by GLOBAL_LOAD_ASYNC_TO_LDS_B128; one barrier
// per 32-wide k-step; stage s+1 DMA overlaps stage s WMMAs.
// OOB patch rows read a pre-zeroed pad region so every lane always issues.
// ---------------------------------------------------------------------------
__global__ __launch_bounds__(256) void k_conv4x4_wmma(
    const _Float16* __restrict__ in,             // [NB, 14>>shift, 14>>shift, 256]
    const _Float16* __restrict__ wt,             // [16][256 co][256 ci]
    const float* __restrict__ bias, _Float16* __restrict__ out,
    const _Float16* __restrict__ zpad, int shift) {
    __shared__ __align__(16) _Float16 As[2][64 * 32];
    __shared__ __align__(16) _Float16 Bs[2][128 * 32];

    const int t = threadIdx.x;
    const int m0 = blockIdx.x * 64;
    const int nb0 = blockIdx.y * 128;
    const int Hin = 14 >> shift;

    const int wave = t >> 5, lane = t & 31;
    const int hl = lane >> 4, l16 = lane & 15;
    const int mw = (wave & 3) * 16, nw = (wave >> 2) * 64;

    // A staging: thread -> (pixel p, 16B chunk q)
    const int p = t >> 2, q = t & 3;
    const int m = m0 + p;
    const int bb = m / 196, rm = m % 196;
    const int oy = rm / 14, ox = rm % 14;

    const unsigned ldsA = LDS_OFF(&As[0][0]);
    const unsigned ldsB = LDS_OFF(&Bs[0][0]);
    const unsigned aoff = (unsigned)(p * 32 + q * 8) * 2u;

    FragC acc[4];
#pragma unroll
    for (int s = 0; s < 4; ++s)
#pragma unroll
        for (int r = 0; r < 8; ++r) acc[s].x[r] = 0.f;

    auto issue = [&](int stage) {
        const unsigned buf = (unsigned)(stage & 1);
        const int tap = stage >> 3, kb = (stage & 7) << 5;
        const int ky = tap >> 2, kx = tap & 3;
        const int iy = oy + ky - 1, ix = ox + kx - 1;     // SAME pad_lo = 1
        const bool valid = (iy >= 0) && (iy < 14) && (ix >= 0) && (ix < 14);
        const _Float16* ga =
            valid ? in + (((long)bb * Hin + (iy >> shift)) * Hin + (ix >> shift)) * 256 +
                        kb + q * 8
                  : zpad;
        async_ld_b128(ldsA + buf * 4096u + aoff, ga);
        const _Float16* wtap = wt + (long)tap * 65536;
#pragma unroll
        for (int i = 0; i < 2; ++i) {
            const int ii = t + i * 256;
            const int nl = ii >> 2, qq = ii & 3;
            const _Float16* gb = wtap + (long)(nb0 + nl) * 256 + kb + qq * 8;
            async_ld_b128(ldsB + buf * 8192u + (unsigned)(nl * 32 + qq * 8) * 2u, gb);
        }
    };

    issue(0);
    for (int stage = 0; stage < 128; ++stage) {
        wait_async0();          // own stage-s tiles have landed in LDS
        __syncthreads();        // everyone's landed; everyone done with buf^1
        if (stage + 1 < 128) issue(stage + 1);   // DMA next tile during compute
        const int buf = stage & 1;
        FragAB a;
#pragma unroll
        for (int j = 0; j < 8; ++j)   // A 16x32 fragment layout per ISA table
            a.u[j] = *(const u32*)(&As[buf][(mw + l16) * 32 + 2 * (j & 3) +
                                            16 * (j >> 2) + 8 * hl]);
#pragma unroll
        for (int s = 0; s < 4; ++s) {
            FragAB bf;
            const int n = nw + s * 16 + l16;
#pragma unroll
            for (int j = 0; j < 8; ++j)   // B 32x16 fragment layout per ISA table (matches A)
                bf.u[j] = *(const u32*)(&Bs[buf][n * 32 + 2 * (j & 3) +
                                                 16 * (j >> 2) + 8 * hl]);
            acc[s].f = __builtin_amdgcn_wmma_f32_16x16x32_f16(
                false, a.h, false, bf.h, (short)0, acc[s].f, false, false);
        }
    }
#pragma unroll
    for (int s = 0; s < 4; ++s) {
        const int n = nb0 + nw + s * 16 + l16;
        const float bv = bias[n];
#pragma unroll
        for (int r = 0; r < 8; ++r) {
            const int mm = m0 + mw + r + 8 * hl;
            out[(long)mm * 256 + n] = (_Float16)fmaxf(acc[s].x[r] + bv, 0.f);
        }
    }
}

// ---------------------------------------------------------------------------
// Generic WMMA GEMM: C[M,N] = A[M,K]f16 @ Bt[N,K]f16 + bias. M%64==0, K%32==0.
// Same async double-buffered pipeline; N guarded via the zero pad region.
// ---------------------------------------------------------------------------
__global__ __launch_bounds__(256) void k_gemm_wmma(
    const _Float16* __restrict__ A, const _Float16* __restrict__ Bt,
    const float* __restrict__ bias, void* __restrict__ outv,
    const _Float16* __restrict__ zpad, int M, int N, int K, int out_f16) {
    __shared__ __align__(16) _Float16 As[2][64 * 32];
    __shared__ __align__(16) _Float16 Bs[2][128 * 32];

    const int t = threadIdx.x;
    const int m0 = blockIdx.x * 64;
    const int nb0 = blockIdx.y * 128;
    const int wave = t >> 5, lane = t & 31;
    const int hl = lane >> 4, l16 = lane & 15;
    const int mw = (wave & 3) * 16, nw = (wave >> 2) * 64;
    const int p = t >> 2, q = t & 3;
    const long arow = (long)(m0 + p) * K;

    const unsigned ldsA = LDS_OFF(&As[0][0]);
    const unsigned ldsB = LDS_OFF(&Bs[0][0]);
    const unsigned aoff = (unsigned)(p * 32 + q * 8) * 2u;

    FragC acc[4];
#pragma unroll
    for (int s = 0; s < 4; ++s)
#pragma unroll
        for (int r = 0; r < 8; ++r) acc[s].x[r] = 0.f;

    const int S = K >> 5;
    auto issue = [&](int stage) {
        const unsigned buf = (unsigned)(stage & 1);
        const int kb = stage << 5;
        async_ld_b128(ldsA + buf * 4096u + aoff, A + arow + kb + q * 8);
#pragma unroll
        for (int i = 0; i < 2; ++i) {
            const int ii = t + i * 256;
            const int nl = ii >> 2, qq = ii & 3;
            const _Float16* gb =
                (nb0 + nl < N) ? Bt + (long)(nb0 + nl) * K + kb + qq * 8 : zpad;
            async_ld_b128(ldsB + buf * 8192u + (unsigned)(nl * 32 + qq * 8) * 2u, gb);
        }
    };

    issue(0);
    for (int stage = 0; stage < S; ++stage) {
        wait_async0();
        __syncthreads();
        if (stage + 1 < S) issue(stage + 1);
        const int buf = stage & 1;
        FragAB a;
#pragma unroll
        for (int j = 0; j < 8; ++j)
            a.u[j] = *(const u32*)(&As[buf][(mw + l16) * 32 + 2 * (j & 3) +
                                            16 * (j >> 2) + 8 * hl]);
#pragma unroll
        for (int s = 0; s < 4; ++s) {
            FragAB bf;
            const int n = nw + s * 16 + l16;
#pragma unroll
            for (int j = 0; j < 8; ++j)
                bf.u[j] = *(const u32*)(&Bs[buf][n * 32 + 2 * (j & 3) +
                                                 16 * (j >> 2) + 8 * hl]);
            acc[s].f = __builtin_amdgcn_wmma_f32_16x16x32_f16(
                false, a.h, false, bf.h, (short)0, acc[s].f, false, false);
        }
    }
#pragma unroll
    for (int s = 0; s < 4; ++s) {
        const int n = nb0 + nw + s * 16 + l16;
        if (n >= N) continue;
        const float bv = bias[n];
#pragma unroll
        for (int r = 0; r < 8; ++r) {
            const int mm = m0 + mw + r + 8 * hl;
            const float v = acc[s].x[r] + bv;
            if (out_f16)
                ((_Float16*)outv)[(long)mm * N + n] = (_Float16)v;
            else
                ((float*)outv)[(long)mm * N + n] = v;
        }
    }
}

// ---------------------------------------------------------------------------
// 2x2 maxpool on f16 [NB,14,14,256] -> [NB,7,7,256]
// ---------------------------------------------------------------------------
__global__ __launch_bounds__(256) void k_pool2(const _Float16* __restrict__ in,
                                               _Float16* __restrict__ out,
                                               int total) {
    int i = blockIdx.x * 256 + threadIdx.x;
    if (i >= total) return;
    int c = i & 255;
    int r = i >> 8;
    int px = r % 7;
    r /= 7;
    int py = r % 7;
    int b = r / 7;
    long base = (((long)b * 14 + 2 * py) * 14 + 2 * px) * 256 + c;
    float v0 = (float)in[base], v1 = (float)in[base + 256];
    float v2 = (float)in[base + 14 * 256], v3 = (float)in[base + 14 * 256 + 256];
    out[i] = (_Float16)fmaxf(fmaxf(v0, v1), fmaxf(v2, v3));
}

// ---------------------------------------------------------------------------
// Squared-distance argmin over 256 SOM embeddings + build f16 latent batch
// zcat = [z_e (rows 0..255); z_q (rows 256..511)]. Block = batch row.
// ---------------------------------------------------------------------------
__global__ __launch_bounds__(256) void k_argmin_zcat(
    const float* __restrict__ z_e, const float* __restrict__ emb,
    _Float16* __restrict__ zcat) {
    const int b = blockIdx.x;
    const int k = threadIdx.x;
    __shared__ float zb[64];
    __shared__ float dist[256];
    __shared__ int idx[256];
    if (k < 64) zb[k] = z_e[b * 64 + k];
    __syncthreads();
    float d = 0.f;
#pragma unroll 8
    for (int j = 0; j < 64; ++j) {
        float t = zb[j] - emb[k * 64 + j];
        d += t * t;
    }
    dist[k] = d;
    idx[k] = k;
    __syncthreads();
    for (int s = 128; s > 0; s >>= 1) {
        if (k < s && dist[k + s] < dist[k]) {
            dist[k] = dist[k + s];
            idx[k] = idx[k + s];
        }
        __syncthreads();
    }
    const int kmin = idx[0];
    if (k < 64) {
        zcat[(long)b * 64 + k] = (_Float16)zb[k];
        zcat[(long)(256 + b) * 64 + k] = (_Float16)emb[kmin * 64 + k];
    }
}

// ---------------------------------------------------------------------------
// deconv2: virtual 2x upsample of [512,14,14,256] then 4x4 conv to 1 channel,
// + bias + sigmoid. Block = output pixel, thread = channel, LDS reduction.
// ---------------------------------------------------------------------------
__global__ __launch_bounds__(256) void k_deconv2_sigmoid(
    const _Float16* __restrict__ d1, const float* __restrict__ w,
    const float* __restrict__ bias, float* __restrict__ out) {
    const long pix = blockIdx.x;                 // n*784 + oy*28 + ox
    const int n = (int)(pix / 784);
    const int rem = (int)(pix % 784);
    const int oy = rem / 28, ox = rem % 28;
    const int c = threadIdx.x;
    float acc = 0.f;
#pragma unroll
    for (int ky = 0; ky < 4; ++ky) {
        int iy = oy + ky - 1;
        if (iy < 0 || iy >= 28) continue;
#pragma unroll
        for (int kx = 0; kx < 4; ++kx) {
            int ix = ox + kx - 1;
            if (ix < 0 || ix >= 28) continue;
            float v = (float)d1[(((long)n * 14 + (iy >> 1)) * 14 + (ix >> 1)) * 256 + c];
            acc += v * w[(ky * 4 + kx) * 256 + c];
        }
    }
    __shared__ float red[256];
    red[c] = acc;
    __syncthreads();
    for (int s = 128; s > 0; s >>= 1) {
        if (c < s) red[c] += red[c + s];
        __syncthreads();
    }
    if (c == 0) {
        float z = red[0] + bias[0];
        out[pix] = 1.f / (1.f + __expf(-z));
    }
}

// ---------------------------------------------------------------------------
// Host-side orchestration
// ---------------------------------------------------------------------------
extern "C" void kernel_launch(void* const* d_in, const int* in_sizes, int n_in,
                              void* d_out, int out_size, void* d_ws,
                              size_t ws_size, hipStream_t stream) {
    (void)in_sizes; (void)n_in; (void)out_size; (void)ws_size;
    const float* x     = (const float*)d_in[0];
    const float* c1w   = (const float*)d_in[1];
    const float* c1b   = (const float*)d_in[2];
    const float* c2w   = (const float*)d_in[3];
    const float* c2b   = (const float*)d_in[4];
    const float* enc_w = (const float*)d_in[5];
    const float* enc_b = (const float*)d_in[6];
    const float* emb   = (const float*)d_in[7];
    const float* dec_w = (const float*)d_in[8];
    const float* dec_b = (const float*)d_in[9];
    const float* d1w   = (const float*)d_in[10];
    const float* d1b   = (const float*)d_in[11];
    const float* d2w   = (const float*)d_in[12];
    const float* d2b   = (const float*)d_in[13];
    float* out = (float*)d_out;

    char* ws = (char*)d_ws;
    size_t off = 0;
    auto carve = [&](size_t bytes) -> void* {
        void* p = ws + off;
        off = (off + bytes + 255) & ~(size_t)255;
        return p;
    };
    _Float16* a1     = (_Float16*)carve(256L * 196 * 256 * 2);   // conv1 pooled
    _Float16* c2o    = (_Float16*)carve(256L * 196 * 256 * 2);   // conv2 pre-pool
    _Float16* a2     = (_Float16*)carve(256L * 49 * 256 * 2);    // conv2 pooled (h_flat)
    _Float16* w2t    = (_Float16*)carve(16L * 256 * 256 * 2);
    _Float16* d1t    = (_Float16*)carve(16L * 256 * 256 * 2);
    _Float16* encwt  = (_Float16*)carve(64L * 12544 * 2);
    _Float16* decwt  = (_Float16*)carve(12544L * 64 * 2);
    float*    z_e    = (float*)carve(256L * 64 * 4);
    _Float16* zcat   = (_Float16*)carve(512L * 64 * 2);
    _Float16* h_dec  = (_Float16*)carve(512L * 12544 * 2);       // [512,7,7,256]
    _Float16* d1out  = (_Float16*)carve(512L * 196 * 256 * 2);   // [512,14,14,256]
    _Float16* zpad   = (_Float16*)carve(256);                    // zero pad region

    // 1. weight prep + zero pad
    k_zero_pad<<<1, 128, 0, stream>>>(zpad);
    k_w_tap_t<<<4096, 256, 0, stream>>>(c2w, w2t);
    k_w_tap_t<<<4096, 256, 0, stream>>>(d1w, d1t);
    k_enc_wt<<<3136, 256, 0, stream>>>(enc_w, encwt);
    k_dec_wt<<<3136, 256, 0, stream>>>(dec_w, decwt);

    // 2. encoder
    k_conv1_relu_pool<<<256 * 196, 256, 0, stream>>>(x, c1w, c1b, a1);
    k_conv4x4_wmma<<<dim3(256 * 196 / 64, 2), 256, 0, stream>>>(a1, w2t, c2b,
                                                                c2o, zpad, 0);
    k_pool2<<<256 * 49 * 256 / 256, 256, 0, stream>>>(c2o, a2, 256 * 49 * 256);
    k_gemm_wmma<<<dim3(256 / 64, 1), 256, 0, stream>>>(a2, encwt, enc_b, z_e,
                                                       zpad, 256, 64, 12544, 0);

    // 3. SOM quantization + latent batch
    k_argmin_zcat<<<256, 256, 0, stream>>>(z_e, emb, zcat);

    // 4. decoder (batched over e/q paths)
    k_gemm_wmma<<<dim3(512 / 64, 98), 256, 0, stream>>>(zcat, decwt, dec_b,
                                                        h_dec, zpad, 512, 12544,
                                                        64, 1);
    k_conv4x4_wmma<<<dim3(512 * 196 / 64, 2), 256, 0, stream>>>(h_dec, d1t, d1b,
                                                                d1out, zpad, 1);
    k_deconv2_sigmoid<<<512 * 784, 256, 0, stream>>>(d1out, d2w, d2b, out);
}
